// MultiHeadAttention_66151086293378
// MI455X (gfx1250) — matrix operand order repaired and checked by reading, not compiled, not run on hardware
//
#include <hip/hip_runtime.h>

// ---------------- CDNA5 types ----------------
typedef __attribute__((ext_vector_type(16))) __bf16 v16bf;
typedef __attribute__((ext_vector_type(8)))  float  v8f;
typedef __attribute__((ext_vector_type(4)))  unsigned int u32x4;

union Frag { v16bf v; u32x4 q[2]; };

// float -> bf16 bits, round-to-nearest-even (scalar path)
__device__ __forceinline__ unsigned short f2bf(float f) {
  unsigned u = __builtin_bit_cast(unsigned, f);
  return (unsigned short)((u + 0x7FFFu + ((u >> 16) & 1u)) >> 16);
}

// pack two floats -> packed bf16 dword in ONE VALU op (v_cvt_pk_bf16_f32).
// (__builtin_amdgcn_cvt_pk_bf16_f32 is absent on this toolchain - probed.)
__device__ __forceinline__ unsigned cvt_pk_bf16(float a, float b) {
  unsigned r;
  asm("v_cvt_pk_bf16_f32 %0, %1, %2" : "=v"(r) : "v"(a), "v"(b));
  return r;
}

// async global->LDS (bypasses VGPRs, ASYNCcnt-tracked) when available.
// Probed signature: (v4i addrspace(1)*, v4i addrspace(3)*, imm offset, imm cpol)
#if __has_builtin(__builtin_amdgcn_global_load_async_to_lds_b128)
#define HAVE_ASYNC_LDS 1
typedef int v4i_vs __attribute__((vector_size(16)));
typedef __attribute__((address_space(1))) v4i_vs* gv4p;
typedef __attribute__((address_space(3))) v4i_vs* lv4p;
#endif

__device__ __forceinline__ void wait_asynccnt0() {
#if __has_builtin(__builtin_amdgcn_s_wait_asynccnt)
  __builtin_amdgcn_s_wait_asynccnt(0);
#else
  asm volatile("s_wait_asynccnt 0x0" ::: "memory");
#endif
}

// Butterfly reductions across the 16-lane half (rows of a WMMA C-frag live on
// lanes 0-15 / 16-31). ds_swizzle group-of-32: imm = (xor<<10) | and_mask(0x1f)
__device__ __forceinline__ float half_reduce_max(float v) {
  v = fmaxf(v, __builtin_bit_cast(float, __builtin_amdgcn_ds_swizzle(__builtin_bit_cast(int, v), 0x041F)));
  v = fmaxf(v, __builtin_bit_cast(float, __builtin_amdgcn_ds_swizzle(__builtin_bit_cast(int, v), 0x081F)));
  v = fmaxf(v, __builtin_bit_cast(float, __builtin_amdgcn_ds_swizzle(__builtin_bit_cast(int, v), 0x101F)));
  v = fmaxf(v, __builtin_bit_cast(float, __builtin_amdgcn_ds_swizzle(__builtin_bit_cast(int, v), 0x201F)));
  return v;
}
__device__ __forceinline__ float half_reduce_sum(float v) {
  v += __builtin_bit_cast(float, __builtin_amdgcn_ds_swizzle(__builtin_bit_cast(int, v), 0x041F));
  v += __builtin_bit_cast(float, __builtin_amdgcn_ds_swizzle(__builtin_bit_cast(int, v), 0x081F));
  v += __builtin_bit_cast(float, __builtin_amdgcn_ds_swizzle(__builtin_bit_cast(int, v), 0x101F));
  v += __builtin_bit_cast(float, __builtin_amdgcn_ds_swizzle(__builtin_bit_cast(int, v), 0x201F));
  return v;
}

#define EMBED 1024
#define HEADS 16
#define NPOS  2048
#define BATCH 4
#define DK    64
#define LSTR  40   // LDS row stride (bf16 elems): keeps 16B chunk alignment, skews banks

// =====================================================================
// Kernel 1: QKV projection.  [8192 x 1024] fp32  @  [1024 x 3072] fp32 + bias
// Block: 256 thr = 8 waves (4M x 2N), tile 64M x 128N, K-step 32.
// Staging: float4 global loads + v_cvt_pk_bf16_f32 packing; A tile stored as
// one b128 per thread, W tile transposed with packed b32 stores (k-pairs).
// Epilogue scatters bf16 Q,K -> [B,H,N,dk], V -> transposed [B,H,dk,N].
// =====================================================================
__global__ __launch_bounds__(256) void qkv_gemm_kernel(
    const float* __restrict__ X, const float* __restrict__ W,
    const float* __restrict__ bias,
    unsigned short* __restrict__ Qb, unsigned short* __restrict__ Kb,
    unsigned short* __restrict__ Vt)
{
  __shared__ unsigned short As[64 * LSTR];    // A tile, row-major [m][k]
  __shared__ unsigned short Bs[128 * LSTR];   // W tile, transposed [n][k]

  const int tid  = threadIdx.x;
  const int lane = tid & 31;
  const int wave = tid >> 5;
  const int mw   = wave >> 1;        // 0..3
  const int nw   = wave & 1;         // 0..1
  const int m0   = blockIdx.y * 64;
  const int n0   = blockIdx.x * 128;
  const int hi   = (lane < 16) ? 0 : 8;    // A-frag K split per lane half
  const int kb   = (lane < 16) ? 0 : 16;   // unused; B-frag K split is hi, as for A

  v8f acc[4] = {};

  for (int k0 = 0; k0 < EMBED; k0 += 32) {
    __syncthreads();
    {  // stage A: 64x32, each thread converts 8 contiguous f32 -> one b128 LDS store
      int r  = tid >> 2;
      int c0 = (tid & 3) * 8;
      const float4* xp = (const float4*)&X[(m0 + r) * EMBED + k0 + c0];
      float4 x0 = xp[0], x1 = xp[1];
      u32x4 pk;
      pk.x = cvt_pk_bf16(x0.x, x0.y);
      pk.y = cvt_pk_bf16(x0.z, x0.w);
      pk.z = cvt_pk_bf16(x1.x, x1.y);
      pk.w = cvt_pk_bf16(x1.z, x1.w);
      *(u32x4*)&As[r * LSTR + c0] = pk;
    }
#pragma unroll
    for (int tt = 0; tt < 2; ++tt) {  // stage W: 32x128 transposed, k-pair packing
      int rp = (tid >> 5) + tt * 8;   // k row-pair index 0..15
      int c0 = (tid & 31) * 4;        // n col
      float4 w0 = *(const float4*)&W[(k0 + 2 * rp + 0) * (3 * EMBED) + n0 + c0];
      float4 w1 = *(const float4*)&W[(k0 + 2 * rp + 1) * (3 * EMBED) + n0 + c0];
      *(unsigned*)&Bs[(c0 + 0) * LSTR + 2 * rp] = cvt_pk_bf16(w0.x, w1.x);
      *(unsigned*)&Bs[(c0 + 1) * LSTR + 2 * rp] = cvt_pk_bf16(w0.y, w1.y);
      *(unsigned*)&Bs[(c0 + 2) * LSTR + 2 * rp] = cvt_pk_bf16(w0.z, w1.z);
      *(unsigned*)&Bs[(c0 + 3) * LSTR + 2 * rp] = cvt_pk_bf16(w0.w, w1.w);
    }
    if (k0 + 32 < EMBED) {            // prefetch next tiles (global_prefetch_b8)
      __builtin_prefetch(&X[(m0 + (tid >> 2)) * EMBED + k0 + 32], 0, 1);
      __builtin_prefetch(&W[(k0 + 32 + 2 * (tid >> 5)) * (3 * EMBED) + n0 + (tid & 31) * 4], 0, 1);
    }
    __syncthreads();

    Frag fa;                          // A-frag: M = 16*mw + lane%16
    {
      const unsigned short* p = &As[(16 * mw + (lane & 15)) * LSTR];
      fa.q[0] = *(const u32x4*)(p + hi);
      fa.q[1] = *(const u32x4*)(p + 16 + hi);
    }
#pragma unroll
    for (int ns = 0; ns < 4; ++ns) {
      Frag fb;                        // B-frag: N = 64*nw + 16*ns + lane%16
      const unsigned short* p = &Bs[(64 * nw + 16 * ns + (lane & 15)) * LSTR + hi];
      fb.q[0] = *(const u32x4*)(p);
      fb.q[1] = *(const u32x4*)(p + 16);
      acc[ns] = __builtin_amdgcn_wmma_f32_16x16x32_bf16(
          false, fa.v, false, fb.v, (short)0, acc[ns], false, false);
    }
  }

  // epilogue: bias + scatter into Q / K / V^T (bf16)
#pragma unroll
  for (int ns = 0; ns < 4; ++ns) {
#pragma unroll
    for (int r = 0; r < 8; ++r) {
      int m = m0 + 16 * mw + ((lane < 16) ? r : r + 8);
      int n = n0 + 64 * nw + 16 * ns + (lane & 15);
      unsigned short bv = f2bf(acc[ns][r] + bias[n]);
      int b = m >> 11, npos = m & (NPOS - 1);
      int sec = n >> 10, c = n & (EMBED - 1);
      int h = c >> 6, d = c & (DK - 1);
      int bh = b * HEADS + h;
      if (sec == 0)      Qb[(bh * NPOS + npos) * DK + d] = bv;
      else if (sec == 1) Kb[(bh * NPOS + npos) * DK + d] = bv;
      else               Vt[(bh * DK + d) * NPOS + npos] = bv;
    }
  }
}

// =====================================================================
// Kernel 2: causal flash attention, one wave per 16 query rows.
// Key blocks of 32 (= WMMA K).  S via 4 WMMAs, online softmax with
// ds_swizzle row reductions, P relayout C->A through per-wave LDS,
// O += P·V via 4 WMMAs against V^T fragments.
// =====================================================================
__global__ __launch_bounds__(128) void attn_kernel(
    const unsigned short* __restrict__ Qb, const unsigned short* __restrict__ Kb,
    const unsigned short* __restrict__ Vt, unsigned short* __restrict__ AO)
{
  __shared__ unsigned short Plds[4][16 * LSTR];   // per-wave P staging

  const int tid  = threadIdx.x;
  const int lane = tid & 31;
  const int wave = tid >> 5;
  const int bh   = blockIdx.x >> 5;                 // b*H + h
  const int q0   = (blockIdx.x & 31) * 64 + wave * 16;
  const int hi   = (lane < 16) ? 0 : 8;

  const unsigned short* Qh = Qb + (size_t)bh * NPOS * DK;
  const unsigned short* Kh = Kb + (size_t)bh * NPOS * DK;
  const unsigned short* Vh = Vt + (size_t)bh * DK * NPOS;

  Frag fq[2];                                       // Q frags, dk chunks of 32
  {
    const unsigned short* p = Qh + (q0 + (lane & 15)) * DK;
#pragma unroll
    for (int c2 = 0; c2 < 2; ++c2) {
      fq[c2].q[0] = *(const u32x4*)(p + 32 * c2 + hi);
      fq[c2].q[1] = *(const u32x4*)(p + 32 * c2 + 16 + hi);
    }
  }

  float mrow[8], lrow[8];
  v8f o[4] = {};
#pragma unroll
  for (int r = 0; r < 8; ++r) { mrow[r] = -1e30f; lrow[r] = 0.f; }

  const int jmax = (q0 + 15) >> 5;
  for (int j = 0; j <= jmax; ++j) {
    const int key0 = j << 5;

    v8f s[2] = {};
#pragma unroll
    for (int n = 0; n < 2; ++n) {                   // S = Q · K^T (scaled later)
#pragma unroll
      for (int c2 = 0; c2 < 2; ++c2) {
        Frag fk;
        const unsigned short* p =
            Kh + (key0 + 16 * n + (lane & 15)) * DK + 32 * c2 + hi;
        fk.q[0] = *(const u32x4*)(p);
        fk.q[1] = *(const u32x4*)(p + 16);
        s[n] = __builtin_amdgcn_wmma_f32_16x16x32_bf16(
            false, fq[c2].v, false, fk.v, (short)0, s[n], false, false);
      }
    }

    const bool partial = (key0 + 31 > q0);          // causal mask needed?
#pragma unroll
    for (int n = 0; n < 2; ++n)
#pragma unroll
      for (int r = 0; r < 8; ++r) {
        float sv = s[n][r] * 0.125f;                // 1/sqrt(64)
        if (partial) {
          int query = q0 + ((lane < 16) ? r : r + 8);
          int key   = key0 + 16 * n + (lane & 15);
          if (key > query) sv = -1e30f;
        }
        s[n][r] = sv;
      }

    float alpha[8];
#pragma unroll
    for (int r = 0; r < 8; ++r) {                   // online softmax per row
      float mx   = half_reduce_max(fmaxf(s[0][r], s[1][r]));
      float mnew = fmaxf(mrow[r], mx);
      float a    = __expf(mrow[r] - mnew);
      float p0   = __expf(s[0][r] - mnew);
      float p1   = __expf(s[1][r] - mnew);
      s[0][r] = p0; s[1][r] = p1;
      lrow[r] = lrow[r] * a + half_reduce_sum(p0 + p1);
      mrow[r] = mnew;
      alpha[r] = a;
    }
#pragma unroll
    for (int d = 0; d < 4; ++d)
#pragma unroll
      for (int r = 0; r < 8; ++r) o[d][r] *= alpha[r];

    // P: C-layout -> A-fragment layout via per-wave LDS round trip
    unsigned short* pl = &Plds[wave][0];
#pragma unroll
    for (int n = 0; n < 2; ++n)
#pragma unroll
      for (int r = 0; r < 8; ++r) {
        int row = (lane < 16) ? r : r + 8;
        pl[row * LSTR + 16 * n + (lane & 15)] = f2bf(s[n][r]);
      }
    asm volatile("s_wait_dscnt 0x0" ::: "memory");  // same-wave DS ordering
    Frag fp;
    {
      const unsigned short* p = pl + (lane & 15) * LSTR;
      fp.q[0] = *(const u32x4*)(p + hi);
      fp.q[1] = *(const u32x4*)(p + 16 + hi);
    }

#pragma unroll
    for (int d = 0; d < 4; ++d) {                   // O += P · V  (V^T layout)
      Frag fv;
      const unsigned short* p =
          Vh + (16 * d + (lane & 15)) * NPOS + key0 + hi;
      fv.q[0] = *(const u32x4*)(p);
      fv.q[1] = *(const u32x4*)(p + 16);
      o[d] = __builtin_amdgcn_wmma_f32_16x16x32_bf16(
          false, fp.v, false, fv.v, (short)0, o[d], false, false);
    }
  }

  // normalize + write attn output (bf16, [B*N, C] with head-major cols)
  const int b = bh >> 4, h = bh & 15;
#pragma unroll
  for (int r = 0; r < 8; ++r) {
    int   row = q0 + ((lane < 16) ? r : r + 8);
    float inv = 1.0f / lrow[r];
#pragma unroll
    for (int d = 0; d < 4; ++d) {
      int col = h * DK + 16 * d + (lane & 15);
      AO[((size_t)b * NPOS + row) * EMBED + col] = f2bf(o[d][r] * inv);
    }
  }
}

// =====================================================================
// Kernel 3: output projection. bf16[8192x1024] @ bf16(W_o) + b_o -> fp32
// A tile staged via GLOBAL_LOAD_ASYNC_TO_LDS_B128 (ASYNCcnt path).
// =====================================================================
__global__ __launch_bounds__(256) void out_gemm_kernel(
    const unsigned short* __restrict__ A, const float* __restrict__ W,
    const float* __restrict__ bias, float* __restrict__ out)
{
  __shared__ unsigned short As[64 * LSTR];
  __shared__ unsigned short Bs[128 * LSTR];

  const int tid  = threadIdx.x;
  const int lane = tid & 31;
  const int wave = tid >> 5;
  const int mw   = wave >> 1;
  const int nw   = wave & 1;
  const int m0   = blockIdx.y * 64;
  const int n0   = blockIdx.x * 128;
  const int hi   = (lane < 16) ? 0 : 8;
  const int kb   = (lane < 16) ? 0 : 16;

  v8f acc[4] = {};

  for (int k0 = 0; k0 < EMBED; k0 += 32) {
    __syncthreads();
    {  // stage A (already bf16): 64x32, one async 16B chunk per thread
      int r  = tid >> 2;
      int c0 = (tid & 3) * 8;
      const unsigned short* gp = &A[(m0 + r) * EMBED + k0 + c0];
#if defined(HAVE_ASYNC_LDS)
      __builtin_amdgcn_global_load_async_to_lds_b128(
          (gv4p)(void*)gp, (lv4p)(void*)&As[r * LSTR + c0], 0, 0);
#else
      *(u32x4*)&As[r * LSTR + c0] = *(const u32x4*)gp;
#endif
    }
#pragma unroll
    for (int tt = 0; tt < 2; ++tt) {  // stage W_o transposed, k-pair packing
      int rp = (tid >> 5) + tt * 8;
      int c0 = (tid & 31) * 4;
      float4 w0 = *(const float4*)&W[(k0 + 2 * rp + 0) * EMBED + n0 + c0];
      float4 w1 = *(const float4*)&W[(k0 + 2 * rp + 1) * EMBED + n0 + c0];
      *(unsigned*)&Bs[(c0 + 0) * LSTR + 2 * rp] = cvt_pk_bf16(w0.x, w1.x);
      *(unsigned*)&Bs[(c0 + 1) * LSTR + 2 * rp] = cvt_pk_bf16(w0.y, w1.y);
      *(unsigned*)&Bs[(c0 + 2) * LSTR + 2 * rp] = cvt_pk_bf16(w0.z, w1.z);
      *(unsigned*)&Bs[(c0 + 3) * LSTR + 2 * rp] = cvt_pk_bf16(w0.w, w1.w);
    }
    if (k0 + 32 < EMBED)
      __builtin_prefetch(&W[(k0 + 32 + 2 * (tid >> 5)) * EMBED + n0 + (tid & 31) * 4], 0, 1);
#if defined(HAVE_ASYNC_LDS)
    wait_asynccnt0();
#endif
    __syncthreads();

    Frag fa;
    {
      const unsigned short* p = &As[(16 * mw + (lane & 15)) * LSTR];
      fa.q[0] = *(const u32x4*)(p + hi);
      fa.q[1] = *(const u32x4*)(p + 16 + hi);
    }
#pragma unroll
    for (int ns = 0; ns < 4; ++ns) {
      Frag fb;
      const unsigned short* p = &Bs[(64 * nw + 16 * ns + (lane & 15)) * LSTR + hi];
      fb.q[0] = *(const u32x4*)(p);
      fb.q[1] = *(const u32x4*)(p + 16);
      acc[ns] = __builtin_amdgcn_wmma_f32_16x16x32_bf16(
          false, fa.v, false, fb.v, (short)0, acc[ns], false, false);
    }
  }

#pragma unroll
  for (int ns = 0; ns < 4; ++ns)
#pragma unroll
    for (int r = 0; r < 8; ++r) {
      int m = m0 + 16 * mw + ((lane < 16) ? r : r + 8);
      int n = n0 + 64 * nw + 16 * ns + (lane & 15);
      out[(size_t)m * EMBED + n] = acc[ns][r] + bias[n];
    }
}

// =====================================================================
extern "C" void kernel_launch(void* const* d_in, const int* in_sizes, int n_in,
                              void* d_out, int out_size, void* d_ws, size_t ws_size,
                              hipStream_t stream) {
  const float* x     = (const float*)d_in[0];   // [4,2048,1024]
  const float* W_qkv = (const float*)d_in[1];   // [1024,3072]
  const float* b_qkv = (const float*)d_in[2];   // [3072]
  const float* W_o   = (const float*)d_in[3];   // [1024,1024]
  const float* b_o   = (const float*)d_in[4];   // [1024]
  float* out = (float*)d_out;                   // [4,2048,1024] fp32

  // workspace carve (bf16): Q, K, V^T, attn-out  = 4 * 16 MB = 64 MB
  const size_t HELEMS = (size_t)BATCH * HEADS * NPOS * DK;  // 8388608
  unsigned short* Qb = (unsigned short*)d_ws;
  unsigned short* Kb = Qb + HELEMS;
  unsigned short* Vt = Kb + HELEMS;
  unsigned short* AO = Vt + HELEMS;

  // 1) QKV projection: grid 24 x 128 tiles of 128N x 64M
  qkv_gemm_kernel<<<dim3(24, 128), 256, 0, stream>>>(x, W_qkv, b_qkv, Qb, Kb, Vt);

  // 2) attention: (B*H) * (N/64) blocks, 4 waves of 16 query rows each
  attn_kernel<<<dim3(BATCH * HEADS * (NPOS / 64)), 128, 0, stream>>>(Qb, Kb, Vt, AO);

  // 3) output projection: grid 8 x 128
  out_gemm_kernel<<<dim3(8, 128), 256, 0, stream>>>(AO, W_o, b_o, out);
}
